// SelfAttention_80049600463167
// MI455X (gfx1250) — matrix operand order repaired and checked by reading, not compiled, not run on hardware
//
#include <hip/hip_runtime.h>
#include <stdint.h>

typedef __attribute__((ext_vector_type(16))) __bf16 v16bf;
typedef __attribute__((ext_vector_type(8)))  float  v8f;
typedef __attribute__((__vector_size__(16))) int    v4i_t;

#define AS1 __attribute__((address_space(1)))
#define AS3 __attribute__((address_space(3)))

union Frag16 {
    v16bf    v;
    uint4    q[2];
    uint16_t u[16];
};

__device__ __forceinline__ uint16_t f2bf(float f) {
    uint32_t u = __float_as_uint(f);
    uint32_t r = u + 0x7FFFu + ((u >> 16) & 1u);   // round-to-nearest-even
    return (uint16_t)(r >> 16);
}

__device__ __forceinline__ v8f wmma_bf16(v16bf a, v16bf b, v8f c) {
    // (neg_a, A, neg_b, B, c_mod, C, reuse_a, reuse_b)
    return __builtin_amdgcn_wmma_f32_16x16x32_bf16(false, a, false, b, (short)0, c, false, false);
}

__device__ __forceinline__ void wait_lds() {
#if defined(__gfx1250__)
    __builtin_amdgcn_wave_barrier();
#if __has_builtin(__builtin_amdgcn_s_wait_dscnt)
    __builtin_amdgcn_s_wait_dscnt(0);
#else
    asm volatile("s_wait_dscnt 0" ::: "memory");
#endif
    __builtin_amdgcn_wave_barrier();
#endif
}

__device__ __forceinline__ void wait_async() {
#if defined(__gfx1250__)
#if __has_builtin(__builtin_amdgcn_s_wait_asynccnt)
    __builtin_amdgcn_s_wait_asynccnt(0);
#else
    asm volatile("s_wait_asynccnt 0" ::: "memory");
#endif
#endif
}

// 16-byte global -> LDS copy: async (ASYNCcnt-tracked) path when available.
__device__ __forceinline__ void cp_b128(void* lds_dst, const void* gsrc) {
#if defined(__gfx1250__) && __has_builtin(__builtin_amdgcn_global_load_async_to_lds_b128)
    __builtin_amdgcn_global_load_async_to_lds_b128(
        (AS1 v4i_t*)gsrc, (AS3 v4i_t*)lds_dst, 0, 0);
#else
    *(uint4*)lds_dst = *(const uint4*)gsrc;
#endif
}

// ---------------------------------------------------------------- fp32 -> bf16
__global__ void cvt_f32_bf16(const float* __restrict__ src,
                             uint16_t* __restrict__ dst, int n) {
    int i = blockIdx.x * blockDim.x + threadIdx.x;
    int stride = gridDim.x * blockDim.x;
    for (; i < n; i += stride) dst[i] = f2bf(src[i]);
}

// ---------------------------------------------------------------- tiled GEMM
// C[M,N] = A[M,K] * W[N,K]^T   (both bf16, row-major, K-contiguous)
// MODE 0: scatter-store qkv: Q (pre-scaled by 1/8) / K per-head row-major, V transposed
// MODE 1: plain f32 store to outF [M,N]
#define GEMM_LDST 40   // halves per LDS row (32 + 8 pad); 80 B = 5*16 keeps 16B alignment

template <int MODE>
__global__ __launch_bounds__(256) void gemm_bf16_kernel(
    const uint16_t* __restrict__ A,
    const uint16_t* __restrict__ W,
    float* __restrict__ outF,
    uint16_t* __restrict__ Qh,
    uint16_t* __restrict__ Kh,
    uint16_t* __restrict__ Vt,
    int M, int N, int K)
{
    __shared__ __align__(16) uint16_t Ash[2][128 * GEMM_LDST];
    __shared__ __align__(16) uint16_t Bsh[2][128 * GEMM_LDST];

    const int tid  = threadIdx.x;
    const int warp = tid >> 5;
    const int lane = tid & 31;
    const int lh   = lane >> 4;   // half-wave select
    const int lm   = lane & 15;

    const int blockM = blockIdx.y * 128;
    const int blockN = blockIdx.x * 128;
    const int wm0 = (warp >> 2) * 64;   // 2 wave-rows (M)
    const int wn0 = (warp & 3) * 32;    // 4 wave-cols (N)

    // stage one 128x32 A-tile + 128x32 W-tile: 512 16B chunks each, 2/thread
    auto stage = [&](int buf, int kb) {
#pragma unroll
        for (int i = 0; i < 2; ++i) {
            int c   = tid + 256 * i;
            int row = c >> 2;
            int col = (c & 3) << 3;
            cp_b128(&Ash[buf][row * GEMM_LDST + col],
                    &A[(size_t)(blockM + row) * K + kb + col]);
            cp_b128(&Bsh[buf][row * GEMM_LDST + col],
                    &W[(size_t)(blockN + row) * K + kb + col]);
        }
    };

    const v8f vzero = {0.f, 0.f, 0.f, 0.f, 0.f, 0.f, 0.f, 0.f};
    v8f acc[4][2];
#pragma unroll
    for (int i = 0; i < 4; ++i)
#pragma unroll
        for (int j = 0; j < 2; ++j) acc[i][j] = vzero;

    stage(0, 0);
    int buf = 0;
    for (int kb = 0; kb < K; kb += 32) {
        wait_async();            // our buf-tile chunks are in LDS
        __syncthreads();         // everyone's chunks are in LDS; prev compute done
        if (kb + 32 < K) stage(buf ^ 1, kb + 32);   // prefetch next tile

        Frag16 af[4], bw[2];
#pragma unroll
        for (int ms = 0; ms < 4; ++ms) {
            const uint16_t* p = &Ash[buf][(wm0 + ms * 16 + lm) * GEMM_LDST + 8 * lh];
            af[ms].q[0] = *(const uint4*)p;          // K = 8*lh + 0..7
            af[ms].q[1] = *(const uint4*)(p + 16);   // K = 16 + 8*lh + 0..7
        }
#pragma unroll
        for (int ns = 0; ns < 2; ++ns) {
            const uint16_t* p = &Bsh[buf][(wn0 + ns * 16 + lm) * GEMM_LDST + 8 * lh];
            bw[ns].q[0] = *(const uint4*)p;          // K = 8*lh + 0..7
            bw[ns].q[1] = *(const uint4*)(p + 16);   // K = 16 + 8*lh + 0..7
        }
#pragma unroll
        for (int ms = 0; ms < 4; ++ms)
#pragma unroll
            for (int ns = 0; ns < 2; ++ns)
                acc[ms][ns] = wmma_bf16(af[ms].v, bw[ns].v, acc[ms][ns]);
        buf ^= 1;
    }

    // epilogue: C layout — VGPR v holds (row v + 8*lh, col lm) of each 16x16 tile
#pragma unroll
    for (int ms = 0; ms < 4; ++ms) {
#pragma unroll
        for (int ns = 0; ns < 2; ++ns) {
            const int n = blockN + wn0 + ns * 16 + lm;
            if (MODE == 1) {
#pragma unroll
                for (int v = 0; v < 8; ++v) {
                    int m = blockM + wm0 + ms * 16 + lh * 8 + v;
                    outF[(size_t)m * N + n] = acc[ms][ns][v];
                }
            } else {
#pragma unroll
                for (int v = 0; v < 8; ++v) {
                    int m  = blockM + wm0 + ms * 16 + lh * 8 + v;
                    int b  = m >> 11;          // row of x -> (batch, seq)
                    int lq = m & 2047;
                    if (n < 1024) {            // Q: [b,h,L,64], pre-scaled by 1/8
                        int hh = n >> 6, d = n & 63;
                        Qh[((size_t)(b * 16 + hh) * 2048 + lq) * 64 + d] =
                            f2bf(acc[ms][ns][v] * 0.125f);
                    } else if (n < 2048) {     // K: [b,h,L,64]
                        int nn = n - 1024;
                        int hh = nn >> 6, d = nn & 63;
                        Kh[((size_t)(b * 16 + hh) * 2048 + lq) * 64 + d] =
                            f2bf(acc[ms][ns][v]);
                    } else {                   // V transposed: [b,h,64,L]
                        int nn = n - 2048;
                        int hh = nn >> 6, d = nn & 63;
                        Vt[((size_t)(b * 16 + hh) * 64 + d) * 2048 + lq] =
                            f2bf(acc[ms][ns][v]);
                    }
                }
            }
        }
    }
}

// ---------------------------------------------------------------- flash attention
// One wave per *pair* of 16-query tiles (32 queries): K/V fragments are loaded
// once per 32-key chunk and feed both tiles (18 WMMAs per chunk). Streaming
// softmax keeps row stats per lane after an LDS transpose of the f32 scores;
// the denominator rides the matrix unit as a 5th accumulator (P @ ones).
#define SST   36            // f32 LDS row stride (32 + 4 pad); 144 B = 9*16
#define STILE (16 * SST)    // one 16x32 score tile

__global__ __launch_bounds__(128) void attn_kernel(
    const uint16_t* __restrict__ Qh,
    const uint16_t* __restrict__ Kh,
    const uint16_t* __restrict__ Vt,
    uint16_t* __restrict__ attout)
{
    __shared__ __align__(16) float Ssh[4][2 * STILE];  // per-wave score staging
    __shared__ __align__(16) float Fsh[4][32];         // per-wave rescale factors

    const int warp = threadIdx.x >> 5;
    const int lane = threadIdx.x & 31;
    const int lh   = lane >> 4;
    const int lm   = lane & 15;

    const int wid = blockIdx.x * 4 + warp;   // 0..2047
    const int qp  = wid & 63;                // query-tile pair
    const int h   = (wid >> 6) & 15;
    const int b   = wid >> 10;

    const uint16_t* Qb = Qh + ((size_t)(b * 16 + h) * 2048 + (size_t)qp * 32) * 64;
    const uint16_t* Kb = Kh + (size_t)(b * 16 + h) * 2048 * 64;
    const uint16_t* Vb = Vt + (size_t)(b * 16 + h) * 64 * 2048;

    // Q A-fragments for both tiles (held for whole loop); Q is pre-scaled by 1/8
    Frag16 aq[2][2];
#pragma unroll
    for (int t = 0; t < 2; ++t) {
        const uint16_t* p = Qb + (size_t)(t * 16 + lm) * 64;
        aq[t][0].q[0] = *(const uint4*)(p + 8 * lh);
        aq[t][0].q[1] = *(const uint4*)(p + 16 + 8 * lh);
        aq[t][1].q[0] = *(const uint4*)(p + 32 + 8 * lh);
        aq[t][1].q[1] = *(const uint4*)(p + 48 + 8 * lh);
    }

    Frag16 ones;
#pragma unroll
    for (int i = 0; i < 16; ++i) ones.u[i] = 0x3F80;  // 1.0bf16

    const v8f vzero = {0.f, 0.f, 0.f, 0.f, 0.f, 0.f, 0.f, 0.f};
    v8f o[2][5];                    // per tile: 4 output d-tiles + denominator
#pragma unroll
    for (int t = 0; t < 2; ++t)
#pragma unroll
        for (int f = 0; f < 5; ++f) o[t][f] = vzero;
    float mrun[2] = {-1e30f, -1e30f};   // running max for row lm, per tile

    float* Sw = &Ssh[warp][0];
    float* Fw = &Fsh[warp][0];

    for (int kc = 0; kc < 2048; kc += 32) {
        // ---- K B-fragments, hoisted for memory-level parallelism
        Frag16 bk[4];
        {
            const uint16_t* p = Kb + (size_t)(kc + lm) * 64 + 8 * lh;  // kv tile 0
            bk[0].q[0] = *(const uint4*)p;        bk[0].q[1] = *(const uint4*)(p + 16);
            bk[1].q[0] = *(const uint4*)(p + 32); bk[1].q[1] = *(const uint4*)(p + 48);
            const uint16_t* p2 = p + 16 * 64;                            // kv tile 1
            bk[2].q[0] = *(const uint4*)p2;        bk[2].q[1] = *(const uint4*)(p2 + 16);
            bk[3].q[0] = *(const uint4*)(p2 + 32); bk[3].q[1] = *(const uint4*)(p2 + 48);
        }

        // ---- S = Q K^T for both tiles (Q carries the 1/8 scale)
        v8f s[2][2];
#pragma unroll
        for (int t = 0; t < 2; ++t) {
            s[t][0] = wmma_bf16(aq[t][0].v, bk[0].v, vzero);
            s[t][0] = wmma_bf16(aq[t][1].v, bk[1].v, s[t][0]);
            s[t][1] = wmma_bf16(aq[t][0].v, bk[2].v, vzero);
            s[t][1] = wmma_bf16(aq[t][1].v, bk[3].v, s[t][1]);
        }

        // ---- V B-fragments issued early: overlap global loads with softmax VALU
        Frag16 bv[4];
#pragma unroll
        for (int f = 0; f < 4; ++f) {
            const uint16_t* p = Vb + (size_t)(f * 16 + lm) * 2048 + kc + 8 * lh;
            bv[f].q[0] = *(const uint4*)p;
            bv[f].q[1] = *(const uint4*)(p + 16);
        }

        // ---- C-layout -> row-per-lane transpose through LDS (f32 scores)
#pragma unroll
        for (int t = 0; t < 2; ++t)
#pragma unroll
            for (int v = 0; v < 8; ++v) {
                int row = lh * 8 + v;
                Sw[t * STILE + row * SST + lm]      = s[t][0][v];
                Sw[t * STILE + row * SST + 16 + lm] = s[t][1][v];
            }
        wait_lds();

        // ---- per-tile: row max, flash factor, exp, pack A-fragment
        Frag16 ap[2];
#pragma unroll
        for (int t = 0; t < 2; ++t) {
            union { uint4 q[4]; float f[16]; } r;   // row lm: kv 8*lh+j, 16+8*lh+j
            const float* pr = Sw + t * STILE + lm * SST + 8 * lh;
            r.q[0] = *(const uint4*)pr;
            r.q[1] = *(const uint4*)(pr + 4);
            r.q[2] = *(const uint4*)(pr + 16);
            r.q[3] = *(const uint4*)(pr + 20);

            float tmax = r.f[0];
#pragma unroll
            for (int j = 1; j < 16; ++j) tmax = fmaxf(tmax, r.f[j]);
            tmax = fmaxf(tmax, __shfl_xor(tmax, 16, 32));
            float mnew = fmaxf(mrun[t], tmax);
            float fsc  = __expf(mrun[t] - mnew);
            mrun[t] = mnew;
            if (lh == 0) Fw[t * 16 + lm] = fsc;
#pragma unroll
            for (int j = 0; j < 16; ++j) ap[t].u[j] = f2bf(__expf(r.f[j] - mnew));
        }
        wait_lds();

        // ---- redistribute flash factors to C-layout rows, rescale accumulators
#pragma unroll
        for (int t = 0; t < 2; ++t) {
            union { uint4 q[2]; float f[8]; } fv;   // rows 8*lh + 0..7
            const float* pf = Fw + t * 16 + 8 * lh;
            fv.q[0] = *(const uint4*)pf;
            fv.q[1] = *(const uint4*)(pf + 4);
#pragma unroll
            for (int f = 0; f < 5; ++f)
#pragma unroll
                for (int v = 0; v < 8; ++v) o[t][f][v] *= fv.f[v];
        }
        wait_lds();

        // ---- O += P @ V ; denominator += P @ ones
#pragma unroll
        for (int t = 0; t < 2; ++t) {
#pragma unroll
            for (int f = 0; f < 4; ++f)
                o[t][f] = wmma_bf16(ap[t].v, bv[f].v, o[t][f]);
            o[t][4] = wmma_bf16(ap[t].v, ones.v, o[t][4]);
        }
    }

    // ---- normalize and store (bf16, head-concatenated row-major for proj GEMM)
#pragma unroll
    for (int t = 0; t < 2; ++t)
#pragma unroll
        for (int v = 0; v < 8; ++v) {
            float inv = 1.0f / o[t][4][v];
            int row = qp * 32 + t * 16 + lh * 8 + v;
            uint16_t* orow = attout + ((size_t)b * 2048 + row) * 1024 + h * 64;
#pragma unroll
            for (int f = 0; f < 4; ++f)
                orow[f * 16 + lm] = f2bf(o[t][f][v] * inv);
        }
}

// ---------------------------------------------------------------- launcher
extern "C" void kernel_launch(void* const* d_in, const int* in_sizes, int n_in,
                              void* d_out, int out_size, void* d_ws, size_t ws_size,
                              hipStream_t stream) {
    const float* x     = (const float*)d_in[0];   // [2,2048,1024]
    const float* Wqkv  = (const float*)d_in[1];   // [3072,1024]
    const float* Wproj = (const float*)d_in[2];   // [1024,1024]
    float* out = (float*)d_out;                   // [2,2048,1024]

    uint16_t* ws     = (uint16_t*)d_ws;           // ~48 MB of bf16 scratch
    uint16_t* xh     = ws;                        // 4,194,304
    uint16_t* wqkvh  = xh     + (size_t)4194304;  // 3,145,728
    uint16_t* wprojh = wqkvh  + (size_t)3145728;  // 1,048,576
    uint16_t* Qh     = wprojh + (size_t)1048576;  // 4,194,304
    uint16_t* Kh     = Qh     + (size_t)4194304;  // 4,194,304
    uint16_t* Vt     = Kh     + (size_t)4194304;  // 4,194,304
    uint16_t* attout = Vt     + (size_t)4194304;  // 4,194,304

    cvt_f32_bf16<<<1024, 256, 0, stream>>>(x, xh, 4194304);
    cvt_f32_bf16<<<768,  256, 0, stream>>>(Wqkv, wqkvh, 3145728);
    cvt_f32_bf16<<<256,  256, 0, stream>>>(Wproj, wprojh, 1048576);

    // qkv = x @ Wqkv^T : M=4096, N=3072, K=1024
    gemm_bf16_kernel<0><<<dim3(24, 32), 256, 0, stream>>>(
        xh, wqkvh, nullptr, Qh, Kh, Vt, 4096, 3072, 1024);

    // attention: 2048 query-tile pairs, 4 waves/block
    attn_kernel<<<512, 128, 0, stream>>>(Qh, Kh, Vt, attout);

    // out = attout @ Wproj^T : M=4096, N=1024, K=1024
    gemm_bf16_kernel<1><<<dim3(8, 32), 256, 0, stream>>>(
        attout, wprojh, out, nullptr, nullptr, nullptr, 4096, 1024, 1024);
}
